// KP_3DGS_88038239634159
// MI455X (gfx1250) — matrix operand order repaired and checked by reading, not compiled, not run on hardware
//
#include <hip/hip_runtime.h>

// ===========================================================================
// KP-3DGS point completion pipeline for MI455X (gfx1250, wave32, WMMA).
//
// v3: 4x M register tiling in the WMMA GEMMs.  Each wave computes a 64x16
// tile with 4 f32 accumulators; one B fragment per K-step feeds 4 WMMAs
// (independent accumulators -> ILP past the WMMA->WMMA hazard), cutting the
// large-operand (activation) L2 traffic 4x and raising per-wave arithmetic
// intensity to ~410 FLOP/B.  Inner loop: 10x global_load_b128 + 4x v_wmma.
//
// Carried from v2:
//  * bf16 everywhere GEMMs touch: weights pre-padded (Kp mult. of 32),
//    activations stored transposed (S x Kp) so fragments are contiguous 16B
//    runs -> pure b128 loads, zero guards, zero in-loop converts.
//  * GEMM epilogue emits next layer's transposed bf16 operand as packed 16B
//    stores (8 acc rows per lane are consecutive M).
//  * sa_knn conv2 + max-over-16-neighbors fused into the epilogue via
//    16-lane shuffle reduction.
// Fragments use contiguous k order (k = 16*half + s) in A and B alike.
// ===========================================================================

typedef __attribute__((ext_vector_type(16))) __bf16 v16bf;
typedef __attribute__((ext_vector_type(8)))  __bf16 v8bf;
typedef __attribute__((ext_vector_type(8)))  float  v8f;

#define BB    16
#define NKNN  16

__device__ __forceinline__ __bf16 f2bf(float f) {
  unsigned u = __builtin_bit_cast(unsigned, f);
  unsigned r = u + 0x7FFFu + ((u >> 16) & 1u);   // RNE
  unsigned short h = (unsigned short)(r >> 16);
  return __builtin_bit_cast(__bf16, h);
}
__device__ __forceinline__ float bf2f(__bf16 h) {
  unsigned x = ((unsigned)__builtin_bit_cast(unsigned short, h)) << 16;
  return __builtin_bit_cast(float, x);
}

// A fragment: one contiguous 16x bf16 run at p
__device__ __forceinline__ v16bf load_a(const __bf16* p) {
  v8bf lo = *(const v8bf*)p;
  v8bf hi = *(const v8bf*)(p + 8);
  return __builtin_shufflevector(lo, hi, 0,1,2,3,4,5,6,7,8,9,10,11,12,13,14,15);
}
// B fragment: one contiguous 16x bf16 run at p
__device__ __forceinline__ v16bf load_b(const __bf16* p) {
  v8bf lo = *(const v8bf*)p;
  v8bf hi = *(const v8bf*)(p + 8);
  return __builtin_shufflevector(lo, hi, 0,1,2,3,4,5,6,7,8,9,10,11,12,13,14,15);
}

#define WMMA_BF16(a, b, c) \
  __builtin_amdgcn_wmma_f32_16x16x32_bf16(false, (a), false, (b), (short)0, (c), false, false)

// ---------------------------------------------------------------------------
// WMMA GEMM, 64(M) x 16(N) tile per wave.  Wb (Mout x Kp) bf16.
// Xb per-batch (S x Kp) bf16 transposed.  outf optional (B,Mout,S) f32;
// outb optional per-batch (S x Mout) bf16.  act: 0=none 1=relu 2=prelu.
// ---------------------------------------------------------------------------
__global__ __launch_bounds__(32)
void k_wmma_gemm(const __bf16* __restrict__ Wb, const float* __restrict__ bias,
                 const __bf16* __restrict__ X, float* outf, __bf16* outb,
                 int Mout, int Kp, int S, int act,
                 const float* __restrict__ alpha, const float* resid)
{
  const int so = blockIdx.x << 4, mo = blockIdx.y << 6, b = blockIdx.z;
  const int lane = threadIdx.x, half = lane >> 4, l16 = lane & 15;
  const size_t arow = (size_t)16 * Kp;
  const __bf16* pa = Wb + (size_t)(mo + l16) * Kp + (half << 4);
  const __bf16* pb = X + (size_t)b * S * Kp + (size_t)(so + l16) * Kp + (half << 4);
  v8f z = {0.f,0.f,0.f,0.f,0.f,0.f,0.f,0.f};
  v8f acc0 = z, acc1 = z, acc2 = z, acc3 = z;
  for (int kk = 0; kk < Kp; kk += 32) {
    v16bf bfr = load_b(pb + kk);
    acc0 = WMMA_BF16(load_a(pa + kk),            bfr, acc0);
    acc1 = WMMA_BF16(load_a(pa + arow + kk),     bfr, acc1);
    acc2 = WMMA_BF16(load_a(pa + 2 * arow + kk), bfr, acc2);
    acc3 = WMMA_BF16(load_a(pa + 3 * arow + kk), bfr, acc3);
  }
  v8f accs[4] = {acc0, acc1, acc2, acc3};

  const int s = so + l16;
  const float al = (act == 2) ? alpha[0] : 0.f;
#pragma unroll
  for (int t = 0; t < 4; ++t) {
    float res[8];
#pragma unroll
    for (int r = 0; r < 8; ++r) {
      const int mm = mo + t * 16 + r + (half << 3);
      float x = accs[t][r] + bias[mm];
      if (act == 1)      x = fmaxf(x, 0.f);
      else if (act == 2) x = (x >= 0.f) ? x : al * x;
      if (resid) x += resid[((size_t)b * Mout + mm) * S + s];
      res[r] = x;
      if (outf) outf[((size_t)b * Mout + mm) * S + s] = x;
    }
    if (outb) {
      v8bf pk;
#pragma unroll
      for (int r = 0; r < 8; ++r) pk[r] = f2bf(res[r]);
      *(v8bf*)(outb + (size_t)b * S * Mout + (size_t)s * Mout
               + mo + t * 16 + (half << 3)) = pk;
    }
  }
}

// GEMM + fused max over each 16-column tile (== max over the KNN group).
// Of (B,Mout,Mg) f32; Ob optional per-batch (Mg x Mout) bf16 transposed.
__global__ __launch_bounds__(32)
void k_wmma_gemm_max16(const __bf16* __restrict__ Wb, const float* __restrict__ bias,
                       const __bf16* __restrict__ X, float* __restrict__ Of,
                       __bf16* __restrict__ Ob, int Mout, int Kp, int Mg)
{
  const int S = Mg << 4;
  const int so = blockIdx.x << 4, mo = blockIdx.y << 6, b = blockIdx.z;
  const int lane = threadIdx.x, half = lane >> 4, l16 = lane & 15;
  const size_t arow = (size_t)16 * Kp;
  const __bf16* pa = Wb + (size_t)(mo + l16) * Kp + (half << 4);
  const __bf16* pb = X + (size_t)b * S * Kp + (size_t)(so + l16) * Kp + (half << 4);
  v8f z = {0.f,0.f,0.f,0.f,0.f,0.f,0.f,0.f};
  v8f acc0 = z, acc1 = z, acc2 = z, acc3 = z;
  for (int kk = 0; kk < Kp; kk += 32) {
    v16bf bfr = load_b(pb + kk);
    acc0 = WMMA_BF16(load_a(pa + kk),            bfr, acc0);
    acc1 = WMMA_BF16(load_a(pa + arow + kk),     bfr, acc1);
    acc2 = WMMA_BF16(load_a(pa + 2 * arow + kk), bfr, acc2);
    acc3 = WMMA_BF16(load_a(pa + 3 * arow + kk), bfr, acc3);
  }
  v8f accs[4] = {acc0, acc1, acc2, acc3};

#pragma unroll
  for (int t = 0; t < 4; ++t) {
    float vals[8];
#pragma unroll
    for (int r = 0; r < 8; ++r) {
      float x = accs[t][r] + bias[mo + t * 16 + r + (half << 3)];
#pragma unroll
      for (int off = 1; off < 16; off <<= 1)
        x = fmaxf(x, __shfl_xor(x, off, 32));
      vals[r] = x;
    }
    if (l16 == 0) {
#pragma unroll
      for (int r = 0; r < 8; ++r)
        Of[((size_t)b * Mout + mo + t * 16 + r + (half << 3)) * Mg + blockIdx.x] = vals[r];
      if (Ob) {
        v8bf pk;
#pragma unroll
        for (int r = 0; r < 8; ++r) pk[r] = f2bf(vals[r]);
        *(v8bf*)(Ob + (size_t)b * Mg * Mout + (size_t)blockIdx.x * Mout
                 + mo + t * 16 + (half << 3)) = pk;
      }
    }
  }
}

// weights f32 (Mout x K) -> bf16 (Mout x Kp), zero padded
__global__ void k_wconv(const float* __restrict__ W, __bf16* __restrict__ Wb,
                        int K, int Kp, int total)
{
  int gid = blockIdx.x * blockDim.x + threadIdx.x;
  if (gid >= total) return;
  int kp = gid % Kp, o = gid / Kp;
  Wb[gid] = f2bf(kp < K ? W[(size_t)o * K + kp] : 0.f);
}

// ---------------------------------------------------------------------------
// FPS: one workgroup per batch, points + running min-dist in LDS (34 KB).
// ---------------------------------------------------------------------------
#define FPS_T 256
__global__ __launch_bounds__(FPS_T)
void k_fps(const float* __restrict__ pts, int N, int npoint, int* __restrict__ out_idx)
{
  __shared__ float sx[2048], sy[2048], sz[2048], dist[2048];
  __shared__ float rmax[FPS_T];
  __shared__ int   rarg[FPS_T];
  const int b = blockIdx.x, t = threadIdx.x;
  const float* P = pts + (size_t)b * 3 * N;
  for (int n = t; n < N; n += FPS_T) {
    sx[n] = P[n]; sy[n] = P[N + n]; sz[n] = P[2 * N + n];
    dist[n] = 1e10f;
  }
  if (t == 0) out_idx[(size_t)b * npoint] = 0;
  __syncthreads();
  int cur = 0;
  for (int i = 1; i < npoint; ++i) {
    const float cx = sx[cur], cy = sy[cur], cz = sz[cur];
    float best = -1.f; int bi = 0;
    for (int n = t; n < N; n += FPS_T) {
      float dx = sx[n] - cx, dy = sy[n] - cy, dz = sz[n] - cz;
      float d = fminf(dist[n], dx * dx + dy * dy + dz * dz);
      dist[n] = d;
      if (d > best) { best = d; bi = n; }
    }
    rmax[t] = best; rarg[t] = bi;
    __syncthreads();
    for (int s = FPS_T / 2; s > 0; s >>= 1) {
      if (t < s) {
        if (rmax[t + s] > rmax[t] ||
            (rmax[t + s] == rmax[t] && rarg[t + s] < rarg[t])) {
          rmax[t] = rmax[t + s]; rarg[t] = rarg[t + s];
        }
      }
      __syncthreads();
    }
    cur = rarg[0];
    if (t == 0) out_idx[(size_t)b * npoint + i] = cur;
    __syncthreads();
  }
}

// KNN (16 smallest sq-dists, ties -> lower index), one thread per center.
__global__ void k_knn(const float* __restrict__ pts, const float* __restrict__ ctr,
                      int N, int M, int* __restrict__ idx)
{
  int gid = blockIdx.x * blockDim.x + threadIdx.x;
  if (gid >= BB * M) return;
  int b = gid / M, m = gid % M;
  const float* X  = pts + (size_t)b * 3 * N;
  const float* Cc = ctr + (size_t)b * 3 * M;
  float cx = Cc[m], cy = Cc[M + m], cz = Cc[2 * M + m];
  float bd[NKNN]; int bi[NKNN];
  for (int j = 0; j < NKNN; ++j) { bd[j] = 3.4e38f; bi[j] = 0; }
  for (int n = 0; n < N; ++n) {
    float dx = X[n] - cx, dy = X[N + n] - cy, dz = X[2 * N + n] - cz;
    float d = dx * dx + dy * dy + dz * dz;
    if (d < bd[NKNN - 1]) {
      int j = NKNN - 1;
      while (j > 0 && d < bd[j - 1]) { bd[j] = bd[j - 1]; bi[j] = bi[j - 1]; --j; }
      bd[j] = d; bi[j] = n;
    }
  }
  int* out = idx + ((size_t)b * M + m) * NKNN;
  for (int j = 0; j < NKNN; ++j) out[j] = bi[j];
}

__global__ void k_gather3(const float* __restrict__ src, const int* __restrict__ fidx,
                          int N, int M, float* __restrict__ dst)
{
  int gid = blockIdx.x * blockDim.x + threadIdx.x;
  if (gid >= BB * 3 * M) return;
  int m = gid % M, t = gid / M, c = t % 3, b = t / 3;
  dst[gid] = src[((size_t)b * 3 + c) * N + fidx[(size_t)b * M + m]];
}

// feat bf16, per-batch rows (M*16 x Kp): [0..2]=centered xyz, [3..3+C)=pts, pad 0
__global__ void k_sa_feat(const float* __restrict__ xyz, const float* __restrict__ pts,
                          const float* __restrict__ nxyz, const int* __restrict__ idx,
                          int N, int M, int C, int Kp, __bf16* __restrict__ feat)
{
  int gid = blockIdx.x * blockDim.x + threadIdx.x;
  if (gid >= BB * M * 16) return;
  int t = gid >> 4, m = t % M, b = t / M;
  int src = idx[(size_t)gid];
  const float* X = xyz + (size_t)b * 3 * N;
  __bf16* F = feat + (size_t)gid * Kp;
  F[0] = f2bf(X[src]         - nxyz[((size_t)b * 3 + 0) * M + m]);
  F[1] = f2bf(X[N + src]     - nxyz[((size_t)b * 3 + 1) * M + m]);
  F[2] = f2bf(X[2 * N + src] - nxyz[((size_t)b * 3 + 2) * M + m]);
  const float* P = pts + (size_t)b * C * N;
  for (int c = 0; c < C; ++c) F[3 + c] = f2bf(P[(size_t)c * N + src]);
  for (int c = 3 + C; c < Kp; ++c) F[c] = f2bf(0.f);
}

// pos_rel bf16, per-batch rows (N*16 x 32)
__global__ void k_posrel(const float* __restrict__ pos, const int* __restrict__ idx,
                         int N, __bf16* __restrict__ pr)
{
  int gid = blockIdx.x * blockDim.x + threadIdx.x;
  if (gid >= BB * N * 16) return;
  int u = gid >> 4, n = u % N, b = u / N;
  int src = idx[(size_t)gid];
  const float* P = pos + (size_t)b * 3 * N;
  __bf16* F = pr + (size_t)gid * 32;
  F[0] = f2bf(P[n] - P[src]);
  F[1] = f2bf(P[N + n] - P[N + src]);
  F[2] = f2bf(P[2 * N + n] - P[2 * N + src]);
#pragma unroll
  for (int c = 3; c < 32; ++c) F[c] = f2bf(0.f);
}

// t bf16 (rows (N*16) x 64): q[b,c,n] - key[b,c,idx] + pe[s][c]
__global__ void k_tbuild(const float* __restrict__ q, const float* __restrict__ keyf,
                         const __bf16* __restrict__ pe, const int* __restrict__ idx,
                         int N, __bf16* __restrict__ t)
{
  int gid = blockIdx.x * blockDim.x + threadIdx.x;
  if (gid >= BB * N * 16 * 64) return;
  int c = gid & 63, u = gid >> 6;
  int n = (u >> 4) % N, b = (u >> 4) / N;
  int src = idx[(size_t)u];
  float val = q[((size_t)b * 64 + c) * N + n]
            - keyf[((size_t)b * 64 + c) * N + src] + bf2f(pe[(size_t)gid]);
  t[gid] = f2bf(val);
}

// softmax over k (stride-64 strip of 16 in the transposed bf16 layout)
__global__ void k_softmax16(__bf16* __restrict__ a, int N)
{
  int gid = blockIdx.x * blockDim.x + threadIdx.x;
  if (gid >= BB * 64 * N) return;
  int n = gid % N, u = gid / N, c = u % 64, b = u / 64;
  __bf16* p = a + ((size_t)b * N * 16 + (size_t)n * 16) * 64 + c;
  float v[16];
  float mx = -3.4e38f;
#pragma unroll
  for (int k = 0; k < 16; ++k) { v[k] = bf2f(p[k * 64]); mx = fmaxf(mx, v[k]); }
  float s = 0.f;
#pragma unroll
  for (int k = 0; k < 16; ++k) { v[k] = expf(v[k] - mx); s += v[k]; }
  float inv = 1.f / s;
#pragma unroll
  for (int k = 0; k < 16; ++k) p[k * 64] = f2bf(v[k] * inv);
}

// agg bf16 (rows N x 64): sum_k attn * (v[b,c,idx] + pe)
__global__ void k_agg(const __bf16* __restrict__ attn, const float* __restrict__ vf,
                      const __bf16* __restrict__ pe, const int* __restrict__ idx,
                      int N, __bf16* __restrict__ agg)
{
  int gid = blockIdx.x * blockDim.x + threadIdx.x;
  if (gid >= BB * 64 * N) return;
  int n = gid % N, u = gid / N, c = u % 64, b = u / 64;
  const int* ip = idx + ((size_t)b * N + n) * 16;
  size_t base = ((size_t)b * N * 16 + (size_t)n * 16) * 64 + c;
  const float* V = vf + ((size_t)b * 64 + c) * N;
  float s = 0.f;
#pragma unroll
  for (int k = 0; k < 16; ++k)
    s += bf2f(attn[base + (size_t)k * 64]) * (V[ip[k]] + bf2f(pe[base + (size_t)k * 64]));
  agg[((size_t)b * N + n) * 64 + c] = f2bf(s);
}

// feat3 bf16 (rows 512 x 544) = [l2_xyz | l2_pts | pad]
__global__ void k_cat_sa3(const float* __restrict__ xyz, const float* __restrict__ pts,
                          __bf16* __restrict__ f)
{
  int gid = blockIdx.x * blockDim.x + threadIdx.x;
  if (gid >= BB * 512 * 544) return;
  int c = gid % 544, u = gid / 544, s = u % 512, b = u / 512;
  float v = 0.f;
  if (c < 3)        v = xyz[((size_t)b * 3 + c) * 512 + s];
  else if (c < 515) v = pts[((size_t)b * 512 + (c - 3)) * 512 + s];
  f[gid] = f2bf(v);
}

// featg bf16 (rows 512 x 1536) = [l2_pts | broadcast(l3)]
__global__ void k_cat_ge(const float* __restrict__ pts, const float* __restrict__ l3,
                         __bf16* __restrict__ f)
{
  int gid = blockIdx.x * blockDim.x + threadIdx.x;
  if (gid >= BB * 512 * 1536) return;
  int c = gid % 1536, u = gid / 1536, s = u % 512, b = u / 512;
  float v = (c < 512) ? pts[((size_t)b * 512 + c) * 512 + s]
                      : l3[(size_t)b * 1024 + (c - 512)];
  f[gid] = f2bf(v);
}

__global__ void k_maxS(const float* __restrict__ x, int Mout, int S, float* __restrict__ o)
{
  int gid = blockIdx.x * blockDim.x + threadIdx.x;
  if (gid >= BB * Mout) return;
  const float* p = x + (size_t)gid * S;
  float m = p[0];
  for (int s = 1; s < S; ++s) m = fmaxf(m, p[s]);
  o[gid] = m;
}

__global__ void k_meanS(const float* __restrict__ x, int Mout, int S, float* __restrict__ o)
{
  int gid = blockIdx.x * blockDim.x + threadIdx.x;
  if (gid >= BB * Mout) return;
  const float* p = x + (size_t)gid * S;
  float s = 0.f;
  for (int i = 0; i < S; ++i) s += p[i];
  o[gid] = s / (float)S;
}

__global__ void k_out_partial(const float* __restrict__ partial, const int* __restrict__ fidx,
                              float* __restrict__ kp_all)
{
  int gid = blockIdx.x * blockDim.x + threadIdx.x;
  if (gid >= BB * 256) return;
  int b = gid / 256, m = gid % 256;
  int src = fidx[gid];
  for (int c = 0; c < 3; ++c)
    kp_all[((size_t)b * 6656 + m) * 3 + c] = partial[((size_t)b * 3 + c) * 2048 + src];
}

__global__ void k_chol_sample(const float* __restrict__ kpg, const float* __restrict__ noise,
                              float* __restrict__ kp_all, float* __restrict__ means)
{
  int gid = blockIdx.x * blockDim.x + threadIdx.x;
  if (gid >= BB * 64) return;
  int b = gid / 64, kp = gid % 64;
  const float* g = kpg + (size_t)b * 576 + (size_t)kp * 9;
  float m0 = g[0], m1 = g[1], m2 = g[2];
  float ce[6];
  for (int j = 0; j < 6; ++j) {
    float v = fabsf(g[3 + j]);
    ce[j] = (v == 0.f) ? 1e-6f : v;
  }
  float c00 = ce[0] * ce[0] + 1e-6f;
  float c10 = ce[3] * ce[0];
  float c11 = ce[3] * ce[3] + ce[1] * ce[1] + 1e-6f;
  float c20 = ce[4] * ce[0];
  float c21 = ce[4] * ce[3] + ce[5] * ce[1];
  float c22 = ce[4] * ce[4] + ce[5] * ce[5] + ce[2] * ce[2] + 1e-6f;
  float l11 = sqrtf(c00);
  float l21 = c10 / l11, l31 = c20 / l11;
  float l22 = sqrtf(fmaxf(c11 - l21 * l21, 0.f));
  float l32 = (c21 - l31 * l21) / l22;
  float l33 = sqrtf(fmaxf(c22 - l31 * l31 - l32 * l32, 0.f));
  means[(size_t)gid * 3 + 0] = m0;
  means[(size_t)gid * 3 + 1] = m1;
  means[(size_t)gid * 3 + 2] = m2;
  for (int s = 0; s < 100; ++s) {
    const float* nz = noise + (((size_t)b * 100 + s) * 64 + kp) * 3;
    float n0 = nz[0], n1 = nz[1], n2 = nz[2];
    float* o = kp_all + ((size_t)b * 6656 + 256 + s * 64 + kp) * 3;
    o[0] = m0 + l11 * n0;
    o[1] = m1 + l21 * n0 + l22 * n1;
    o[2] = m2 + l31 * n0 + l32 * n1 + l33 * n2;
  }
}

// ---------------------------------------------------------------------------
// Host orchestration
// ---------------------------------------------------------------------------
static inline int cdiv(int a, int b) { return (a + b - 1) / b; }

struct Bump {
  char* base; size_t off;
  void* take(size_t bytes) {
    size_t a = (off + 255) & ~(size_t)255;
    void* p = base + a; off = a + bytes; return p;
  }
};

static void gemm(hipStream_t st, const __bf16* Wb, const float* bias, const __bf16* X,
                 float* outf, __bf16* outb, int Mout, int Kp, int S, int batch, int act,
                 const float* alpha = nullptr, const float* resid = nullptr)
{
  dim3 g(S / 16, Mout / 64, batch), blk(32);
  k_wmma_gemm<<<g, blk, 0, st>>>(Wb, bias, X, outf, outb, Mout, Kp, S, act, alpha, resid);
}

static void gemm_max(hipStream_t st, const __bf16* Wb, const float* bias, const __bf16* X,
                     float* Of, __bf16* Ob, int Mout, int Kp, int Mg, int batch)
{
  dim3 g(Mg, Mout / 64, batch), blk(32);
  k_wmma_gemm_max16<<<g, blk, 0, st>>>(Wb, bias, X, Of, Ob, Mout, Kp, Mg);
}

struct VtW {
  const __bf16 *start, *key, *query, *value, *pos1, *pos2, *attn1, *attn2, *end;
  const float  *start_b, *key_b, *query_b, *value_b, *pos_b1, *pos_b2,
               *attn_b1, *attn_b2, *end_b;
};

// Point-transformer block. pts f32 (B,inc,N) in/out; ptsbf per-batch (N x inc).
static void run_vtrans(hipStream_t st, float* pts, const __bf16* ptsbf, const float* pos,
                       int N, int inc, const VtW& W, Bump& A, int* idx)
{
  const int S16 = N * 16;
  float*  q    = (float*)A.take((size_t)BB * 64 * N * 4);
  float*  keyf = (float*)A.take((size_t)BB * 64 * N * 4);
  float*  v    = (float*)A.take((size_t)BB * 64 * N * 4);
  __bf16* x64  = (__bf16*)A.take((size_t)BB * N * 64 * 2);
  __bf16* pe   = (__bf16*)A.take((size_t)BB * S16 * 64 * 2);
  size_t sub = A.off;
  __bf16* posr = (__bf16*)A.take((size_t)BB * S16 * 32 * 2);
  __bf16* ph   = (__bf16*)A.take((size_t)BB * S16 * 64 * 2);

  k_knn<<<cdiv(BB * N, 256), 256, 0, st>>>(pos, pos, N, N, idx);
  gemm(st, W.start, W.start_b, ptsbf, nullptr, x64, 64, inc, N, BB, 0);
  gemm(st, W.key,   W.key_b,   x64, keyf, nullptr, 64, 64, N, BB, 0);
  gemm(st, W.query, W.query_b, x64, q,    nullptr, 64, 64, N, BB, 0);
  gemm(st, W.value, W.value_b, x64, v,    nullptr, 64, 64, N, BB, 0);
  k_posrel<<<cdiv(BB * N * 16, 256), 256, 0, st>>>(pos, idx, N, posr);
  gemm(st, W.pos1, W.pos_b1, posr, nullptr, ph, 64, 32, S16, BB, 1);  // relu
  gemm(st, W.pos2, W.pos_b2, ph,   nullptr, pe, 64, 64, S16, BB, 0);

  A.off = sub;  // free posr, ph
  __bf16* t    = (__bf16*)A.take((size_t)BB * S16 * 64 * 2);
  __bf16* attn = (__bf16*)A.take((size_t)BB * S16 * 64 * 2);
  __bf16* a1   = (__bf16*)A.take((size_t)S16 * 256 * 2);
  __bf16* aggb = (__bf16*)A.take((size_t)BB * N * 64 * 2);

  k_tbuild<<<cdiv(BB * S16 * 64, 256), 256, 0, st>>>(q, keyf, pe, idx, N, t);
  for (int b = 0; b < BB; ++b) {   // batch-split caps the 256-ch scratch
    gemm(st, W.attn1, W.attn_b1, t + (size_t)b * S16 * 64, nullptr, a1, 256, 64, S16, 1, 1);
    gemm(st, W.attn2, W.attn_b2, a1, nullptr, attn + (size_t)b * S16 * 64, 64, 256, S16, 1, 0);
  }
  k_softmax16<<<cdiv(BB * 64 * N, 256), 256, 0, st>>>(attn, N);
  k_agg<<<cdiv(BB * 64 * N, 256), 256, 0, st>>>(attn, v, pe, idx, N, aggb);
  gemm(st, W.end, W.end_b, aggb, pts, nullptr, inc, 64, N, BB, 0, nullptr, pts); // +resid
}

// ---- input indices (dict insertion order; params flattened sorted-keys) ----
enum {
  IN_PARTIAL = 0, IN_NOISE = 1,
  P_GE1_A = 2, P_GE1_B, P_GE1_W, P_GE2_A, P_GE2_B, P_GE2_W,
  P_GE3_A, P_GE3_B, P_GE3_W,
  P_SA1_B1, P_SA1_B2, P_SA1_W1, P_SA1_W2,
  P_SA2_B1, P_SA2_B2, P_SA2_W1, P_SA2_W2,
  P_SA3_B1, P_SA3_B2, P_SA3_W1, P_SA3_W2,
  P_T1 = 23, P_T2 = 41
};
// t-block leaf order (sorted): attn_b1, attn_b2, attn_w1, attn_w2, end_b,
// end_w, key_b, key_w, pos_b1, pos_b2, pos_w1, pos_w2, query_b, query_w,
// start_b, start_w, value_b, value_w

extern "C" void kernel_launch(void* const* d_in, const int* in_sizes, int n_in,
                              void* d_out, int out_size, void* d_ws, size_t ws_size,
                              hipStream_t stream)
{
  (void)in_sizes; (void)n_in; (void)out_size; (void)ws_size;
  auto F = [&](int i) { return (const float*)d_in[i]; };
  const float* partial = F(IN_PARTIAL);
  const float* noise   = F(IN_NOISE);

  Bump A{(char*)d_ws, 0};

  // ---- persistent buffers ----
  int*   fidx256 = (int*)A.take((size_t)BB * 256 * 4);
  int*   fidx1   = (int*)A.take((size_t)BB * 1024 * 4);
  int*   fidx2   = (int*)A.take((size_t)BB * 512 * 4);
  int*   idxA    = (int*)A.take((size_t)BB * 1024 * 16 * 4);
  int*   idxB    = (int*)A.take((size_t)BB * 1024 * 16 * 4);
  float* l1_xyz  = (float*)A.take((size_t)BB * 3 * 1024 * 4);
  float* l2_xyz  = (float*)A.take((size_t)BB * 3 * 512 * 4);
  float* l1_pts  = (float*)A.take((size_t)BB * 128 * 1024 * 4);
  float* l2_pts  = (float*)A.take((size_t)BB * 512 * 512 * 4);
  __bf16* l1ptsb = (__bf16*)A.take((size_t)BB * 1024 * 128 * 2);
  __bf16* l2ptsb = (__bf16*)A.take((size_t)BB * 512 * 512 * 2);
  float* l3      = (float*)A.take((size_t)BB * 1024 * 4);
  float* kpg     = (float*)A.take((size_t)BB * 576 * 4);

  // ---- bf16 weight staging (one conversion pass per launch) ----
  auto wconv = [&](int pidx, int Mout, int K, int Kp) -> const __bf16* {
    __bf16* wb = (__bf16*)A.take((size_t)Mout * Kp * 2);
    int total = Mout * Kp;
    k_wconv<<<cdiv(total, 256), 256, 0, stream>>>(F(pidx), wb, K, Kp, total);
    return wb;
  };
  const __bf16* w_sa1_1 = wconv(P_SA1_W1, 64, 6, 32);
  const __bf16* w_sa1_2 = wconv(P_SA1_W2, 128, 64, 64);
  const __bf16* w_sa2_1 = wconv(P_SA2_W1, 256, 131, 160);
  const __bf16* w_sa2_2 = wconv(P_SA2_W2, 512, 256, 256);
  const __bf16* w_sa3_1 = wconv(P_SA3_W1, 512, 515, 544);
  const __bf16* w_sa3_2 = wconv(P_SA3_W2, 1024, 512, 512);
  const __bf16* w_ge1   = wconv(P_GE1_W, 512, 1536, 1536);
  const __bf16* w_ge2   = wconv(P_GE2_W, 256, 512, 512);
  const __bf16* w_ge3   = wconv(P_GE3_W, 576, 256, 256);
  auto vtw = [&](int base, int inc) {
    VtW w;
    w.attn_b1 = F(base + 0);  w.attn_b2 = F(base + 1);
    w.attn1 = wconv(base + 2, 256, 64, 64);
    w.attn2 = wconv(base + 3, 64, 256, 256);
    w.end_b = F(base + 4);
    w.end   = wconv(base + 5, inc, 64, 64);
    w.key_b = F(base + 6);
    w.key   = wconv(base + 7, 64, 64, 64);
    w.pos_b1 = F(base + 8);  w.pos_b2 = F(base + 9);
    w.pos1  = wconv(base + 10, 64, 3, 32);
    w.pos2  = wconv(base + 11, 64, 64, 64);
    w.query_b = F(base + 12);
    w.query = wconv(base + 13, 64, 64, 64);
    w.start_b = F(base + 14);
    w.start = wconv(base + 15, 64, inc, inc);
    w.value_b = F(base + 16);
    w.value = wconv(base + 17, 64, 64, 64);
    return w;
  };
  VtW vt1 = vtw(P_T1, 128);
  VtW vt2 = vtw(P_T2, 512);

  const size_t mark = A.off;

  // ---- FPS ----
  k_fps<<<BB, FPS_T, 0, stream>>>(partial, 2048, 256,  fidx256);
  k_fps<<<BB, FPS_T, 0, stream>>>(partial, 2048, 1024, fidx1);

  // ---- SA1: 2048 -> 1024 centers, 6 -> 64 -> 128 ch, fused max16 ----
  {
    A.off = mark;
    __bf16* feat1 = (__bf16*)A.take((size_t)BB * 16384 * 32 * 2);
    __bf16* h1    = (__bf16*)A.take((size_t)BB * 16384 * 64 * 2);
    k_gather3<<<cdiv(BB * 3 * 1024, 256), 256, 0, stream>>>(partial, fidx1, 2048, 1024, l1_xyz);
    k_knn<<<cdiv(BB * 1024, 256), 256, 0, stream>>>(partial, l1_xyz, 2048, 1024, idxA);
    k_sa_feat<<<cdiv(BB * 1024 * 16, 256), 256, 0, stream>>>(partial, partial, l1_xyz,
                                                             idxA, 2048, 1024, 3, 32, feat1);
    gemm(stream, w_sa1_1, F(P_SA1_B1), feat1, nullptr, h1, 64, 32, 16384, BB, 1);
    gemm_max(stream, w_sa1_2, F(P_SA1_B2), h1, l1_pts, l1ptsb, 128, 64, 1024, BB);
  }

  // ---- Transformer 1 (N=1024, inc=128) ----
  A.off = mark;
  run_vtrans(stream, l1_pts, l1ptsb, l1_xyz, 1024, 128, vt1, A, idxB);

  // ---- SA2: 1024 -> 512 centers, 131 -> 256 -> 512 ch ----
  {
    A.off = mark;
    __bf16* feat2 = (__bf16*)A.take((size_t)BB * 8192 * 160 * 2);
    __bf16* h2    = (__bf16*)A.take((size_t)BB * 8192 * 256 * 2);
    k_fps<<<BB, FPS_T, 0, stream>>>(l1_xyz, 1024, 512, fidx2);
    k_gather3<<<cdiv(BB * 3 * 512, 256), 256, 0, stream>>>(l1_xyz, fidx2, 1024, 512, l2_xyz);
    k_knn<<<cdiv(BB * 512, 256), 256, 0, stream>>>(l1_xyz, l2_xyz, 1024, 512, idxA);
    k_sa_feat<<<cdiv(BB * 512 * 16, 256), 256, 0, stream>>>(l1_xyz, l1_pts, l2_xyz,
                                                            idxA, 1024, 512, 128, 160, feat2);
    gemm(stream, w_sa2_1, F(P_SA2_B1), feat2, nullptr, h2, 256, 160, 8192, BB, 1);
    gemm_max(stream, w_sa2_2, F(P_SA2_B2), h2, l2_pts, l2ptsb, 512, 256, 512, BB);
  }

  // ---- Transformer 2 (N=512, inc=512) ----
  A.off = mark;
  run_vtrans(stream, l2_pts, l2ptsb, l2_xyz, 512, 512, vt2, A, idxB);

  // ---- SA3 (global): 515 -> 512 -> 1024, max over 512 points ----
  {
    A.off = mark;
    __bf16* feat3 = (__bf16*)A.take((size_t)BB * 512 * 544 * 2);
    __bf16* h3    = (__bf16*)A.take((size_t)BB * 512 * 512 * 2);
    float*  o3    = (float*)A.take((size_t)BB * 1024 * 512 * 4);
    k_cat_sa3<<<cdiv(BB * 512 * 544, 256), 256, 0, stream>>>(l2_xyz, l2_pts, feat3);
    gemm(stream, w_sa3_1, F(P_SA3_B1), feat3, nullptr, h3, 512, 544, 512, BB, 1);
    gemm(stream, w_sa3_2, F(P_SA3_B2), h3, o3, nullptr, 1024, 512, 512, BB, 0);
    k_maxS<<<cdiv(BB * 1024, 256), 256, 0, stream>>>(o3, 1024, 512, l3);
  }

  // ---- Head: 1536 -> 512 -> 256 -> 576 (PReLU), mean over points ----
  {
    A.off = mark;
    __bf16* featg = (__bf16*)A.take((size_t)BB * 512 * 1536 * 2);
    __bf16* g1    = (__bf16*)A.take((size_t)BB * 512 * 512 * 2);
    __bf16* g2    = (__bf16*)A.take((size_t)BB * 512 * 256 * 2);
    float*  g3    = (float*)A.take((size_t)BB * 576 * 512 * 4);
    k_cat_ge<<<cdiv(BB * 512 * 1536, 256), 256, 0, stream>>>(l2_pts, l3, featg);
    gemm(stream, w_ge1, F(P_GE1_B), featg, nullptr, g1, 512, 1536, 512, BB, 2, F(P_GE1_A));
    gemm(stream, w_ge2, F(P_GE2_B), g1,    nullptr, g2, 256, 512,  512, BB, 2, F(P_GE2_A));
    gemm(stream, w_ge3, F(P_GE3_B), g2,    g3, nullptr, 576, 256,  512, BB, 2, F(P_GE3_A));
    k_meanS<<<cdiv(BB * 576, 256), 256, 0, stream>>>(g3, 576, 512, kpg);
  }

  // ---- Outputs: kp_all (B,6656,3) then means (B,64,3) ----
  float* kp_all = (float*)d_out;
  float* means  = kp_all + (size_t)BB * 6656 * 3;
  k_out_partial<<<cdiv(BB * 256, 256), 256, 0, stream>>>(partial, fidx256, kp_all);
  k_chol_sample<<<cdiv(BB * 64, 256), 256, 0, stream>>>(kpg, noise, kp_all, means);
}
